// RNNSampler_77584289235497
// MI455X (gfx1250) — matrix operand order repaired and checked by reading, not compiled, not run on hardware
//
#include <hip/hip_runtime.h>

// ---------------------------------------------------------------------------
// RNN expression sampler for MI455X (gfx1250, wave32, WMMA).
// One persistent kernel: each block owns 16 sequences and iterates all 64
// autoregressive steps internally. The per-step logits GEMM (16x512 @ 512x64)
// runs on v_wmma_f32_16x16x32_bf16; B fragments (packed Wp) live in registers
// (128 VGPRs/wave, loaded once); A fragments stream through a 4-deep rotating
// register pipeline, with sched_group_barrier directives pinning the
// DS-read / WMMA interleave so LDS latency hides under the WMMAs.
// ---------------------------------------------------------------------------

#define VOCAB    64
#define HID      512
#define MAXL     64
#define MIN_LEN  4
#define NTOT     2048
#define SEQ_PER_BLK 16
#define THREADS  128
#define HSTRIDE  520   // bf16 elements per h row in LDS (pad to spread banks)

typedef __attribute__((ext_vector_type(16))) __bf16   v16bf;
typedef __attribute__((ext_vector_type(8)))  float    v8f;
typedef __attribute__((ext_vector_type(4)))  unsigned u32x4;

union VFrag { v16bf v; u32x4 u[2]; };

__device__ __forceinline__ float rng_uniform(unsigned a, unsigned b) {
    unsigned x = a * 0x9E3779B1u ^ b * 0x85EBCA77u ^ 0x4BEEF123u;
    x ^= x >> 16; x *= 0x7FEB352Du;
    x ^= x >> 15; x *= 0x846CA68Bu;
    x ^= x >> 16;
    return (float)(x >> 8) * (1.0f / 16777216.0f);
}

// ---- prep: hc = h0@Wh + b ; x0h = x0@Wx --------------------------------
__global__ void prep_vec_kernel(const float* __restrict__ hid,
                                const float* __restrict__ Wh,
                                const float* __restrict__ b,
                                const float* __restrict__ x0,
                                const float* __restrict__ Wx,
                                float* __restrict__ hc,
                                float* __restrict__ x0h) {
    int k = blockIdx.x * blockDim.x + threadIdx.x;
    if (k >= HID) return;
    float acc = b[k];
    for (int j = 0; j < HID; ++j) acc += hid[j] * Wh[(size_t)j * HID + k];
    hc[k] = acc;
    float a2 = 0.f;
    for (int r = 0; r < 2 * VOCAB; ++r) a2 += x0[r] * Wx[(size_t)r * HID + k];
    x0h[k] = a2;
}

// ---- prep: pack Wp (512x64 f32) into bf16 WMMA B-fragment order ---------
// idx = (((ntile*16 + kchunk)*32 + lane)*16 + e)
// lane<16:  col = nt*16+lane,    k = kc*32 + e        (K 0..15)
// lane>=16: col = nt*16+lane-16, k = kc*32 + 16 + e   (K 16..31)
__global__ void prep_pack_kernel(const float* __restrict__ Wp,
                                 __bf16* __restrict__ wpack) {
    int idx = blockIdx.x * blockDim.x + threadIdx.x;
    if (idx >= 4 * 16 * 32 * 16) return;
    int e    = idx & 15;
    int lane = (idx >> 4) & 31;
    int kc   = (idx >> 9) & 15;
    int nt   = idx >> 13;
    int col  = nt * 16 + (lane & 15);
    int k    = kc * 32 + ((lane < 16) ? 0 : 16) + e;
    wpack[idx] = (__bf16)Wp[(size_t)k * VOCAB + col];
}

// ---- prep: token class table: bit0=two-arity bit1=one-arity bit2=var ----
__global__ void prep_cls_kernel(const int* __restrict__ two, int n2,
                                const int* __restrict__ one, int n1,
                                const int* __restrict__ var, int nv,
                                int* __restrict__ cls) {
    int v = threadIdx.x;
    if (v >= VOCAB) return;
    int c = 0;
    for (int i = 0; i < n2; ++i) if (two[i] == v) c |= 1;
    for (int i = 0; i < n1; ++i) if (one[i] == v) c |= 2;
    for (int i = 0; i < nv; ++i) if (var[i] == v) c |= 4;
    cls[v] = c;
}

// ---- main persistent sampler kernel -------------------------------------
__global__ __launch_bounds__(THREADS)
void rnn_sampler_main(const float* __restrict__ Wx,
                      const float* __restrict__ bp,
                      const float* __restrict__ hc,
                      const float* __restrict__ x0h,
                      const int*   __restrict__ cls,
                      const __bf16* __restrict__ wpack,
                      int*   __restrict__ o_seq,
                      int*   __restrict__ o_len,
                      float* __restrict__ o_ent,
                      float* __restrict__ o_lp) {
    __shared__ __align__(16) __bf16 s_h[SEQ_PER_BLK * HSTRIDE];
    __shared__ float s_logits[SEQ_PER_BLK][VOCAB];
    __shared__ float s_redA[SEQ_PER_BLK][8];
    __shared__ float s_redB[SEQ_PER_BLK][8];
    __shared__ int   s_seq[SEQ_PER_BLK][MAXL];
    __shared__ float s_hc[HID];
    __shared__ int   s_cls[VOCAB];
    __shared__ float s_bp[VOCAB];
    __shared__ int   s_par[SEQ_PER_BLK], s_sib[SEQ_PER_BLK];
    __shared__ int   s_cnt[SEQ_PER_BLK], s_alive[SEQ_PER_BLK];
    __shared__ int   s_hasvar[SEQ_PER_BLK], s_len[SEQ_PER_BLK];
    __shared__ float s_ent[SEQ_PER_BLK], s_lp[SEQ_PER_BLK];

    const int tid  = threadIdx.x;
    const int blk  = blockIdx.x;
    const int wave = tid >> 5;
    const int lane = tid & 31;

    for (int k = tid; k < HID; k += THREADS) s_hc[k] = hc[k];
    if (tid < VOCAB) { s_cls[tid] = cls[tid]; s_bp[tid] = bp[tid]; }
    if (tid < SEQ_PER_BLK) {
        s_par[tid] = -1; s_sib[tid] = -1;
        s_cnt[tid] = 1;  s_alive[tid] = 1;
        s_hasvar[tid] = 0; s_len[tid] = 0;
        s_ent[tid] = 0.f; s_lp[tid] = 0.f;
    }
    for (int i = tid; i < SEQ_PER_BLK * MAXL; i += THREADS)
        (&s_seq[0][0])[i] = -1;

    // Preload this wave's entire B operand (its 16-column N-tile of Wp for
    // all K=512) into registers: 16 chunks x 8 VGPRs. Stays resident across
    // all 64 steps -> zero VMEM in the hot loop.
    VFrag barr[16];
    #pragma unroll
    for (int kc = 0; kc < 16; ++kc) {
        const __bf16* wp_ = wpack + (((size_t)(wave * 16 + kc) * 32) + lane) * 16;
        barr[kc].u[0] = ((const u32x4*)wp_)[0];
        barr[kc].u[1] = ((const u32x4*)wp_)[1];
    }
    __syncthreads();

    for (int t = 0; t < MAXL; ++t) {
        // ---- Phase 1: h = tanh(Wx[parent] + Wx[64+sibling] + hc) -> bf16
        {
            const int j  = tid >> 3;
            const int kb = (tid & 7) * 64;
            __bf16* hrow = &s_h[j * HSTRIDE];
            if (t == 0) {
                for (int k = 0; k < 64; ++k) {
                    int kk = kb + k;
                    hrow[kk] = (__bf16)tanhf(x0h[kk] + s_hc[kk]);
                }
            } else {
                const int p  = s_par[j];
                const int si = s_sib[j];
                const float* wxp = (p  >= 0) ? (Wx + (size_t)p * HID) : nullptr;
                const float* wxs = (si >= 0) ? (Wx + (size_t)(VOCAB + si) * HID) : nullptr;
                for (int k = 0; k < 64; k += 4) {
                    int kk = kb + k;
                    float4 a = make_float4(0.f, 0.f, 0.f, 0.f);
                    float4 c = make_float4(0.f, 0.f, 0.f, 0.f);
                    if (wxp) a = *(const float4*)(wxp + kk);
                    if (wxs) c = *(const float4*)(wxs + kk);
                    hrow[kk + 0] = (__bf16)tanhf(a.x + c.x + s_hc[kk + 0]);
                    hrow[kk + 1] = (__bf16)tanhf(a.y + c.y + s_hc[kk + 1]);
                    hrow[kk + 2] = (__bf16)tanhf(a.z + c.z + s_hc[kk + 2]);
                    hrow[kk + 3] = (__bf16)tanhf(a.w + c.w + s_hc[kk + 3]);
                }
            }
        }
        __syncthreads();

        // ---- Phase 2: logits tile via v_wmma_f32_16x16x32_bf16
        // 4-deep rotating A pipeline + sched_group_barrier directives:
        // prologue of 8 DS reads (4 fragments), then alternate 1 WMMA with
        // 2 DS reads -> >=2 fragments of loads in flight under every WMMA.
        {
            v8f acc = {0.f, 0.f, 0.f, 0.f, 0.f, 0.f, 0.f, 0.f};
            const int row  = lane & 15;
            const int koff = (lane < 16) ? 0 : 16;  // A-fragment K sub-offset
            const __bf16* hbase = &s_h[row * HSTRIDE];
            VFrag af[4];
            #pragma unroll
            for (int q = 0; q < 4; ++q) {
                af[q].u[0] = *(const u32x4*)(hbase + q * 32 + koff);
                af[q].u[1] = *(const u32x4*)(hbase + q * 32 + 8 + koff);
            }
            #pragma unroll
            for (int kc = 0; kc < 16; ++kc) {
                acc = __builtin_amdgcn_wmma_f32_16x16x32_bf16(
                    false, af[kc & 3].v, false, barr[kc].v, (short)0, acc, false, false);
                if (kc + 4 < 16) {
                    af[kc & 3].u[0] = *(const u32x4*)(hbase + (kc + 4) * 32 + koff);
                    af[kc & 3].u[1] = *(const u32x4*)(hbase + (kc + 4) * 32 + 8 + koff);
                }
            }
#if __has_builtin(__builtin_amdgcn_sched_group_barrier)
            // Pin the pipeline shape: 8 DS reads up front, then {WMMA, 2 DS}.
            // Masks: 0x100 = DS read, 0x008 = MFMA/WMMA.
            __builtin_amdgcn_sched_group_barrier(0x100, 8, 0);
            #pragma unroll
            for (int it = 0; it < 12; ++it) {
                __builtin_amdgcn_sched_group_barrier(0x008, 1, 0);
                __builtin_amdgcn_sched_group_barrier(0x100, 2, 0);
            }
            #pragma unroll
            for (int it = 0; it < 4; ++it)
                __builtin_amdgcn_sched_group_barrier(0x008, 1, 0);
#endif
            const int m0  = (lane < 16) ? 0 : 8;
            const int col = wave * 16 + (lane & 15);
            #pragma unroll
            for (int r = 0; r < 8; ++r) s_logits[m0 + r][col] = acc[r];
        }
        __syncthreads();

        // ---- Phase 3: mask + renormalized softmax + entropy partials
        {
            const int j   = tid >> 3;
            const int g   = tid & 7;
            const int v0  = g * 8;
            const int cnt = s_cnt[j];
            const int tot = t + cnt;
            const int hv  = s_hasvar[j];
            float l[8]; int keep[8];
            float lmax = -3.4e38f;
            #pragma unroll
            for (int q = 0; q < 8; ++q) {
                int v = v0 + q;
                int c = s_cls[v];
                int istwo = c & 1, isone = (c >> 1) & 1, isvar = (c >> 2) & 1;
                int iszero = !(istwo | isone);
                int iscon  = iszero && !isvar;
                int kp = 1;
                if (iszero && tot < MIN_LEN)       kp = 0;
                if (istwo  && (tot + 2 > MAXL))    kp = 0;
                if (isone  && (tot + 1 > MAXL))    kp = 0;
                if (iscon  && (!hv && cnt == 1))   kp = 0;
                keep[q] = kp;
                l[q] = s_logits[j][v] + s_bp[v];
                if (kp && l[q] > lmax) lmax = l[q];
            }
            s_redA[j][g] = lmax;
            __syncthreads();
            float m = s_redA[j][0];
            #pragma unroll
            for (int q = 1; q < 8; ++q) m = fmaxf(m, s_redA[j][q]);
            float e[8]; float psum = 0.f;
            #pragma unroll
            for (int q = 0; q < 8; ++q) {
                e[q] = keep[q] ? expf(l[q] - m) : 0.f;
                psum += e[q];
            }
            s_redB[j][g] = psum;
            __syncthreads();
            float S = 0.f;
            #pragma unroll
            for (int q = 0; q < 8; ++q) S += s_redB[j][q];
            S = fmaxf(S, 1e-30f);
            const float inv = 1.f / S;
            float entp = 0.f;
            #pragma unroll
            for (int q = 0; q < 8; ++q) {
                float p = e[q] * inv;
                s_logits[j][v0 + q] = p;       // overwrite logits with probs
                if (p > 0.f) entp -= p * logf(p);
            }
            __syncthreads();                    // WAR on s_redA
            s_redA[j][g] = entp;
        }
        __syncthreads();

        // ---- Phase 4: sample token + state machine (1 thread / sequence)
        if (tid < SEQ_PER_BLK) {
            const int j = tid;
            float entj = 0.f;
            #pragma unroll
            for (int g = 0; g < 8; ++g) entj += s_redA[j][g];

            const unsigned gseq = (unsigned)(blk * SEQ_PER_BLK + j);
            const float u = rng_uniform(gseq, (unsigned)t);
            float c = 0.f, ptok = 0.f; int tok = -1;
            for (int v = 0; v < VOCAB; ++v) {
                float pv = s_logits[j][v];
                c += pv;
                if (tok < 0 && pv > 0.f && u < c) { tok = v; ptok = pv; }
            }
            if (tok < 0) {
                for (int v = VOCAB - 1; v >= 0; --v)
                    if (s_logits[j][v] > 0.f) { tok = v; ptok = s_logits[j][v]; break; }
            }
            if (tok < 0) { tok = 0; ptok = 1.f; }
            const float lp = logf(ptok);

            const int cc  = s_cls[tok];
            int cnt = s_cnt[j] - 1 + ((cc & 1) ? 2 : 0) + ((cc & 2) ? 1 : 0);
            int alive = s_alive[j] && (cnt > 0);
            s_cnt[j] = cnt; s_alive[j] = alive;
            if (cc & 4) s_hasvar[j] = 1;
            s_seq[j][t] = tok;
            if (alive) { s_ent[j] += entj; s_lp[j] += lp; s_len[j] += 1; }

            // parent/sibling: if the new token is an operator it IS the
            // parent (reference override) -> skip the scan entirely.
            int par, sib;
            if (cc & 3) {
                par = tok; sib = -1;
            } else {
                par = -1; sib = -1;
                int cscan = 0;
                for (int k = 0; k <= t; ++k) {
                    int i  = t - k;
                    int tk = s_seq[j][i];
                    int cv = s_cls[tk];
                    cscan = cscan - 1 + ((cv & 1) ? 2 : 0) + ((cv & 2) ? 1 : 0);
                    if (cscan == 0) {
                        par = tk;
                        sib = (i == t) ? -1 : s_seq[j][i + 1];
                        break;              // first hit wins; rest is dead work
                    }
                }
            }
            s_par[j] = par; s_sib[j] = sib;

            // Prefetch next step's Wx rows toward the WGP (global_prefetch_b8)
            if (par >= 0) {
                const char* rp = (const char*)(Wx + (size_t)par * HID);
                __builtin_prefetch(rp, 0, 3);
                __builtin_prefetch(rp + 1024, 0, 3);
            }
            if (sib >= 0) {
                const char* rs = (const char*)(Wx + (size_t)(VOCAB + sib) * HID);
                __builtin_prefetch(rs, 0, 3);
                __builtin_prefetch(rs + 1024, 0, 3);
            }
        }
        __syncthreads();
    }

    // ---- write outputs: seq(int), lengths(int), ent(f32), lp(f32)
    {
        const int j = tid >> 3, g = tid & 7;
        const unsigned gseq = (unsigned)(blk * SEQ_PER_BLK + j);
        #pragma unroll
        for (int q = 0; q < 8; ++q)
            o_seq[(size_t)gseq * MAXL + g * 8 + q] = s_seq[j][g * 8 + q];
    }
    if (tid < SEQ_PER_BLK) {
        const unsigned gseq = (unsigned)(blk * SEQ_PER_BLK + tid);
        o_len[gseq] = s_len[tid] + 1;
        o_ent[gseq] = s_ent[tid];
        o_lp[gseq]  = s_lp[tid];
    }
}

// ---------------------------------------------------------------------------
extern "C" void kernel_launch(void* const* d_in, const int* in_sizes, int n_in,
                              void* d_out, int out_size, void* d_ws, size_t ws_size,
                              hipStream_t stream) {
    (void)n_in; (void)out_size; (void)ws_size;

    const float* input_init  = (const float*)d_in[0];   // (1, 128)
    const float* hidden_init = (const float*)d_in[1];   // (1, 512)
    const float* Wx          = (const float*)d_in[2];   // (128, 512)
    const float* Wh          = (const float*)d_in[3];   // (512, 512)
    const float* b           = (const float*)d_in[4];   // (512,)
    const float* Wp          = (const float*)d_in[5];   // (512, 64)
    const float* bp          = (const float*)d_in[6];   // (64,)
    const int*   two_ar      = (const int*)d_in[7];
    const int*   one_ar      = (const int*)d_in[8];
    const int*   var_tok     = (const int*)d_in[9];
    // d_in[10] == n (device scalar); reference constant N=2048 used for grid.

    char* ws = (char*)d_ws;
    float*  hc    = (float*)(ws + 0);          // 512 f32
    float*  x0h   = (float*)(ws + 2048);       // 512 f32
    int*    cls   = (int*)  (ws + 4096);       // 64 i32
    __bf16* wpack = (__bf16*)(ws + 4352);      // 32768 bf16 = 64KB

    prep_vec_kernel<<<2, 256, 0, stream>>>(hidden_init, Wh, b, input_init, Wx, hc, x0h);
    prep_pack_kernel<<<128, 256, 0, stream>>>(Wp, wpack);
    prep_cls_kernel<<<1, 64, 0, stream>>>(two_ar, in_sizes[7],
                                          one_ar, in_sizes[8],
                                          var_tok, in_sizes[9], cls);

    char* out = (char*)d_out;
    int*   o_seq = (int*)out;
    int*   o_len = (int*)  (out + (size_t)NTOT * MAXL * 4);
    float* o_ent = (float*)(out + (size_t)NTOT * MAXL * 4 + (size_t)NTOT * 4);
    float* o_lp  = (float*)(out + (size_t)NTOT * MAXL * 4 + (size_t)NTOT * 8);

    rnn_sampler_main<<<NTOT / SEQ_PER_BLK, THREADS, 0, stream>>>(
        Wx, bp, hc, x0h, cls, wpack, o_seq, o_len, o_ent, o_lp);
}
